// PrototypeContextualizer_20624432956329
// MI455X (gfx1250) — compile-verified, matrix operand order checked by reading, not run on hardware
//
#include <hip/hip_runtime.h>

// PrototypeContextualizer: out = X + softmax((norm(X) norm(X)^T)/sqrt(D)) @ X
// N=8192 rows, D=2048 features, fp32 in/out.
//
// Pipeline (f16 WMMA for the two big GEMMs, async global->LDS double buffering):
//   k_rnorm   : rnorm[i] = 1/max(||X[i]||, 1e-12)
//   k_convert : Fh = fp16(X * rnorm) row-major;  Vt = fp16(X)^T (D x N)
//   k_scores  : S = Fh Fh^T / sqrt(D); P = exp(S) (fp16); lsum[i] = sum_j exp(S_ij)
//               (|S| <= 1/sqrt(D) ~ 0.022, so exp without max-subtraction is exact softmax)
//   k_apply   : out = X + (P @ V)/lsum
//
// Workspace layout (needs ~192.1 MiB of d_ws):
//   rnorm: N*4 | lsum: N*4 | Fh: N*D*2 | Vt: N*D*2 | P: N*N*2

typedef __attribute__((ext_vector_type(16))) _Float16 v16h;
typedef __attribute__((ext_vector_type(8)))  _Float16 v8h;
typedef __attribute__((ext_vector_type(8)))  float    v8f;

#define NROWS 8192
#define NDIM  2048
#define BM 128
#define BN 128
#define BK 64
#define LDSS 72   // LDS row stride in halves (64 + 8 pad); 144B = 9*16B keeps alignment

// ---- CDNA5 async global->LDS copy (ASYNCcnt path), 16B per lane ----
__device__ __forceinline__ void async_copy_b128(const _Float16* lds_dst,
                                                const _Float16* gsrc) {
  const unsigned lds_off = (unsigned)(size_t)lds_dst;  // low 32 bits = LDS offset
  asm volatile("global_load_async_to_lds_b128 %0, %1, off"
               :: "v"(lds_off), "v"(gsrc) : "memory");
}
__device__ __forceinline__ void wait_async0() {
  asm volatile("s_wait_asynccnt 0" ::: "memory");
}

// Stage a BMxBK f16 tile pair into LDS asynchronously.
// Per thread: 4 chunks of 16B per matrix; row = tid/8 + j*32, col = (tid&7)*8.
__device__ __forceinline__ void stage_tiles_async(
    _Float16* ldsA, _Float16* ldsB,
    const _Float16* __restrict__ gA, const _Float16* __restrict__ gB,
    size_t lda, size_t ldb, int tid) {
  const int col  = (tid & 7) * 8;
  const int row0 = tid >> 3;
#pragma unroll
  for (int j = 0; j < 4; ++j) {
    const int r = row0 + j * 32;
    async_copy_b128(ldsA + r * LDSS + col, gA + (size_t)r * lda + col);
    async_copy_b128(ldsB + r * LDSS + col, gB + (size_t)r * ldb + col);
  }
}

// Load one 16x32 f16 A/B fragment (k-step ks of a BK=64 tile) from LDS.
// ISA 16-bit layout: lane&15 = matrix row; lanes>=16 shift K by 8;
// halves 0..7 -> K = koff+0..7, halves 8..15 -> K = koff+16..23.
__device__ __forceinline__ v16h load_frag(const _Float16* lds, int rowBase,
                                          int ks, int lane) {
  const int r    = rowBase + (lane & 15);
  const int koff = (lane >> 4) << 3;
  const _Float16* p = lds + r * LDSS + ks * 32 + koff;
  v8h lo = *(const v8h*)(p);        // K = koff + 0..7
  v8h hi = *(const v8h*)(p + 16);   // K = koff + 16..23
  v16h out;
#pragma unroll
  for (int i = 0; i < 8; ++i) { out[i] = lo[i]; out[i + 8] = hi[i]; }
  return out;
}

// ---------------- kernel 1a: reciprocal row norms ----------------
__global__ __launch_bounds__(256)
void k_rnorm(const float* __restrict__ X, float* __restrict__ rnorm) {
  __shared__ float red[8];
  const int tid = threadIdx.x;
  const float* row = X + (size_t)blockIdx.x * NDIM;
  float ss = 0.0f;
#pragma unroll
  for (int i = 0; i < 2; ++i) {
    float4 v = *(const float4*)(row + tid * 8 + i * 4);
    ss += v.x * v.x + v.y * v.y + v.z * v.z + v.w * v.w;
  }
  ss += __shfl_xor(ss, 16, 32);
  ss += __shfl_xor(ss,  8, 32);
  ss += __shfl_xor(ss,  4, 32);
  ss += __shfl_xor(ss,  2, 32);
  ss += __shfl_xor(ss,  1, 32);
  if ((tid & 31) == 0) red[tid >> 5] = ss;
  __syncthreads();
  if (tid == 0) {
    float t = 0.0f;
#pragma unroll
    for (int w = 0; w < 8; ++w) t += red[w];
    float n = fmaxf(sqrtf(t), 1e-12f);
    rnorm[blockIdx.x] = 1.0f / n;
  }
}

// ---------------- kernel 1b: convert + transpose ----------------
__global__ __launch_bounds__(256)
void k_convert(const float* __restrict__ X, const float* __restrict__ rnorm,
               _Float16* __restrict__ Fh, _Float16* __restrict__ Vt) {
  __shared__ float tile[64 * 65];
  const int tid   = threadIdx.x;
  const int rbase = blockIdx.y * 64;   // prototype row
  const int cbase = blockIdx.x * 64;   // feature dim
#pragma unroll
  for (int i = 0; i < 16; ++i) {
    const int linear = tid + i * 256;
    const int r = linear >> 6;
    const int c = linear & 63;
    const float v = X[(size_t)(rbase + r) * NDIM + cbase + c];
    tile[r * 65 + c] = v;
    Fh[(size_t)(rbase + r) * NDIM + cbase + c] = (_Float16)(v * rnorm[rbase + r]);
  }
  __syncthreads();
#pragma unroll
  for (int i = 0; i < 16; ++i) {
    const int linear = tid + i * 256;
    const int d = linear >> 6;
    const int n = linear & 63;
    Vt[(size_t)(cbase + d) * NROWS + rbase + n] = (_Float16)tile[n * 65 + d];
  }
}

// ---------------- kernel 2: scores GEMM + exp + row-sums ----------------
__global__ __launch_bounds__(256, 2)
void k_scores(const _Float16* __restrict__ F, _Float16* __restrict__ P,
              float* __restrict__ lsum) {
  __shared__ __align__(16) _Float16 As[2][BM * LDSS];
  __shared__ __align__(16) _Float16 Bs[2][BN * LDSS];
  const int tid   = threadIdx.x;
  const int lane  = tid & 31;
  const int wave  = tid >> 5;
  const int waveM = wave & 1;   // 2 waves along M (64 rows each)
  const int waveN = wave >> 1;  // 4 waves along N (32 cols each)
  const int mbase = blockIdx.y * BM;
  const int nbase = blockIdx.x * BN;
  const float scale = 0.022097086912079608f; // 1/sqrt(2048)

  const _Float16* gA = F + (size_t)mbase * NDIM;
  const _Float16* gB = F + (size_t)nbase * NDIM;

  v8f acc[4][2] = {};

  // prologue: stage first tile into buffer 0
  stage_tiles_async(As[0], Bs[0], gA, gB, NDIM, NDIM, tid);

  int buf = 0;
  for (int k0 = 0; k0 < NDIM; k0 += BK) {
    wait_async0();
    __syncthreads();
    if (k0 + BK < NDIM)
      stage_tiles_async(As[buf ^ 1], Bs[buf ^ 1],
                        gA + k0 + BK, gB + k0 + BK, NDIM, NDIM, tid);
#pragma unroll
    for (int ks = 0; ks < 2; ++ks) {
      v16h bfr[2];
#pragma unroll
      for (int tn = 0; tn < 2; ++tn)
        bfr[tn] = load_frag(Bs[buf], waveN * 32 + tn * 16, ks, lane);
#pragma unroll
      for (int tm = 0; tm < 4; ++tm) {
        v16h afr = load_frag(As[buf], waveM * 64 + tm * 16, ks, lane);
#pragma unroll
        for (int tn = 0; tn < 2; ++tn)
          acc[tm][tn] = __builtin_amdgcn_wmma_f32_16x16x32_f16(
              false, afr, false, bfr[tn], (short)0, acc[tm][tn], false, false);
      }
    }
    buf ^= 1;
  }

  // epilogue: P = exp(S*scale) (fp16), lsum += row partial sums
  const int lgrp = (lane >> 4) << 3;  // C layout: lanes>=16 -> M+8
  const int lcol = lane & 15;
#pragma unroll
  for (int tm = 0; tm < 4; ++tm) {
#pragma unroll
    for (int r = 0; r < 8; ++r) {
      const int grow = mbase + waveM * 64 + tm * 16 + lgrp + r;
      float ps = 0.0f;
#pragma unroll
      for (int tn = 0; tn < 2; ++tn) {
        const int gcol = nbase + waveN * 32 + tn * 16 + lcol;
        const float p = __expf(acc[tm][tn][r] * scale);
        P[(size_t)grow * NROWS + gcol] = (_Float16)p;
        ps += p;
      }
      ps += __shfl_xor(ps, 1, 32);
      ps += __shfl_xor(ps, 2, 32);
      ps += __shfl_xor(ps, 4, 32);
      ps += __shfl_xor(ps, 8, 32);
      if (lcol == 0) atomicAdd(&lsum[grow], ps);
    }
  }
}

// ---------------- kernel 3: out = X + (P @ V)/lsum ----------------
__global__ __launch_bounds__(256, 2)
void k_apply(const _Float16* __restrict__ P, const _Float16* __restrict__ Vt,
             const float* __restrict__ lsum, const float* __restrict__ X,
             float* __restrict__ out) {
  __shared__ __align__(16) _Float16 As[2][BM * LDSS];
  __shared__ __align__(16) _Float16 Bs[2][BN * LDSS];
  const int tid   = threadIdx.x;
  const int lane  = tid & 31;
  const int wave  = tid >> 5;
  const int waveM = wave & 1;
  const int waveN = wave >> 1;
  const int mbase = blockIdx.y * BM;   // query rows
  const int nbase = blockIdx.x * BN;   // output feature dims

  const _Float16* gA = P  + (size_t)mbase * NROWS;
  const _Float16* gB = Vt + (size_t)nbase * NROWS;  // B column n = contiguous Vt row

  v8f acc[4][2] = {};

  stage_tiles_async(As[0], Bs[0], gA, gB, NROWS, NROWS, tid);

  int buf = 0;
  for (int k0 = 0; k0 < NROWS; k0 += BK) {
    wait_async0();
    __syncthreads();
    if (k0 + BK < NROWS)
      stage_tiles_async(As[buf ^ 1], Bs[buf ^ 1],
                        gA + k0 + BK, gB + k0 + BK, NROWS, NROWS, tid);
#pragma unroll
    for (int ks = 0; ks < 2; ++ks) {
      v16h bfr[2];
#pragma unroll
      for (int tn = 0; tn < 2; ++tn)
        bfr[tn] = load_frag(Bs[buf], waveN * 32 + tn * 16, ks, lane);
#pragma unroll
      for (int tm = 0; tm < 4; ++tm) {
        v16h afr = load_frag(As[buf], waveM * 64 + tm * 16, ks, lane);
#pragma unroll
        for (int tn = 0; tn < 2; ++tn)
          acc[tm][tn] = __builtin_amdgcn_wmma_f32_16x16x32_f16(
              false, afr, false, bfr[tn], (short)0, acc[tm][tn], false, false);
      }
    }
    buf ^= 1;
  }

  const int lgrp = (lane >> 4) << 3;
  const int lcol = lane & 15;
#pragma unroll
  for (int tm = 0; tm < 4; ++tm) {
#pragma unroll
    for (int r = 0; r < 8; ++r) {
      const int grow = mbase + waveM * 64 + tm * 16 + lgrp + r;
      const float inv = 1.0f / lsum[grow];
#pragma unroll
      for (int tn = 0; tn < 2; ++tn) {
        const int gcol = nbase + waveN * 32 + tn * 16 + lcol;
        out[(size_t)grow * NDIM + gcol] =
            X[(size_t)grow * NDIM + gcol] + acc[tm][tn][r] * inv;
      }
    }
  }
}

extern "C" void kernel_launch(void* const* d_in, const int* in_sizes, int n_in,
                              void* d_out, int out_size, void* d_ws, size_t ws_size,
                              hipStream_t stream) {
  const float* X = (const float*)d_in[0];
  float* out = (float*)d_out;

  char* ws = (char*)d_ws;
  float*    rnorm = (float*)ws;
  float*    lsum  = (float*)(ws + (size_t)NROWS * sizeof(float));
  _Float16* Fh    = (_Float16*)(ws + (size_t)2 * NROWS * sizeof(float));
  _Float16* Vt    = Fh + (size_t)NROWS * NDIM;
  _Float16* P     = Vt + (size_t)NROWS * NDIM;

  hipMemsetAsync(lsum, 0, NROWS * sizeof(float), stream);

  k_rnorm  <<<NROWS, 256, 0, stream>>>(X, rnorm);
  k_convert<<<dim3(NDIM / 64, NROWS / 64), 256, 0, stream>>>(X, rnorm, Fh, Vt);
  k_scores <<<dim3(NROWS / BN, NROWS / BM), 256, 0, stream>>>(Fh, P, lsum);
  k_apply  <<<dim3(NDIM / BN, NROWS / BM), 256, 0, stream>>>(P, Vt, lsum, X, out);
}
